// MutualInformation_12704513262346
// MI455X (gfx1250) — matrix operand order repaired and checked by reading, not compiled, not run on hardware
//
#include <hip/hip_runtime.h>
#include <hip/hip_bf16.h>

typedef __attribute__((ext_vector_type(16))) _Float16 v16h;
typedef __attribute__((ext_vector_type(8)))  _Float16 v8h;
typedef __attribute__((ext_vector_type(8)))  float    v8f;

#define NBINS   256
#define NPIX    65536          // 256*256 pixels per image
#define CHUNK   64             // pixels staged in LDS per iteration (2 K-steps)
#define NBLK    32             // n-slices per batch
#define ITERS   (NPIX / NBLK / CHUNK)   // 32
#define PADROW  72             // LDS row length in halfs (144B: 16B aligned, conflict-free stride)
#define KC      (-4.8828125f)  // -0.5 / (0.32*0.32)
#define XSCALE  1275.0f        // 255 * 5
#define EPS     1e-10f

__global__ void mi_zero(float* __restrict__ p, int n) {
    int i = blockIdx.x * blockDim.x + threadIdx.x;
    if (i < n) p[i] = 0.0f;
}

// Grid: (NBLK, 8).  Block: 1024 threads = 32 waves (one full WGP).
// Each block accumulates a partial 256x256 joint histogram (kv1^T @ kv2) for
// one batch over a 2048-pixel slice, plus the marginal column sums.
// Each wave owns a 32(M) x 64(Nj) block = 2 i-tiles x 4 j-tiles: every B
// fragment feeds two back-to-back WMMAs (1.5 ds_load_b128 per WMMA), and all
// fragment loads use immediate DS offsets off just two base-address VGPRs.
__global__ __launch_bounds__(1024)
void mi_joint(const float* __restrict__ in1, const float* __restrict__ in2,
              const float* __restrict__ bins,
              float* __restrict__ joint, float* __restrict__ marg1,
              float* __restrict__ marg2) {
    __shared__ alignas(16) _Float16 lkv1[NBINS * PADROW];  // [bin][n] f16
    __shared__ alignas(16) _Float16 lkv2[NBINS * PADROW];
    __shared__ float lxa[CHUNK];
    __shared__ float lxb[CHUNK];

    const int tid = threadIdx.x;
    const int bb  = blockIdx.y;  // 0..7 (post-concat batch)
    const float* a_src = (bb < 4) ? (in1 + bb * NPIX) : (in2 + (bb - 4) * NPIX);
    const float* b_src = in2 + (bb & 3) * NPIX;
    const int region = blockIdx.x * (NPIX / NBLK);

    // kv-compute mapping: 4 threads per bin, 16 pixels each
    const int bin = tid >> 2;
    const int n0  = (tid & 3) * 16;
    const float binv = bins[bin];

    // WMMA mapping: wave -> 2 i-tiles (rows 32*pi..) x 4 j-tiles (cols 64*pj..)
    const int wave = tid >> 5, lane = tid & 31;
    const int pi   = wave >> 2;          // 0..7 : i-tile pair -> i-tiles 2pi, 2pi+1
    const int pj   = wave & 3;           // 0..3 : j-tile quad -> j-tiles 4pj..4pj+3
    const int m    = lane & 15;          // M row (A) / N col (B)
    const int kbA  = (lane >> 4) * 8;    // A K-base per half-wave (0 or 8)
    const int kbB  = (lane >> 4) * 8;    // B K-base per half-wave (0 or 8)

    // Single base pointer per matrix; all tile/K displacements are constants
    // that fold into the 16-bit DS immediate offset (max used: ~7 KB).
    const _Float16* abase = &lkv1[(pi * 32 + m) * PADROW + kbA];
    const _Float16* bbase = &lkv2[(pj * 64 + m) * PADROW + kbB];

    union Frag { v16h v; v8h h[2]; };

    v8f acc[2][4] = {};
    float s1 = 0.0f, s2 = 0.0f;

    for (int it = 0; it < ITERS; ++it) {
        const int nbase = region + it * CHUNK;
        if (tid < 2 * CHUNK) {
            const float* src = (tid < CHUNK) ? a_src : b_src;
            float*       dst = (tid < CHUNK) ? lxa   : lxb;
            const int    n   = tid & (CHUNK - 1);
            dst[n] = src[nbase + n] * XSCALE;
        }
        __syncthreads();

        // Gaussian kernel values -> LDS (f16); marginal sums ride along.
        _Float16 t1[16], t2[16];
#pragma unroll
        for (int k = 0; k < 16; ++k) {
            float r1 = lxa[n0 + k] - binv;
            float v1 = __expf(r1 * r1 * KC);
            s1 += v1; t1[k] = (_Float16)v1;
            float r2 = lxb[n0 + k] - binv;
            float v2 = __expf(r2 * r2 * KC);
            s2 += v2; t2[k] = (_Float16)v2;
        }
        *(v8h*)&lkv1[bin * PADROW + n0]     = *(v8h*)&t1[0];
        *(v8h*)&lkv1[bin * PADROW + n0 + 8] = *(v8h*)&t1[8];
        *(v8h*)&lkv2[bin * PADROW + n0]     = *(v8h*)&t2[0];
        *(v8h*)&lkv2[bin * PADROW + n0 + 8] = *(v8h*)&t2[8];
        // Pin the marginal sums here so their 16 exp temporaries die at the
        // stores instead of staying live across the whole WMMA section.
        asm volatile("" : "+v"(s1), "+v"(s2));
        __syncthreads();

        // Two K=32 steps over the staged 64 pixels.
#pragma unroll
        for (int ks = 0; ks < 2; ++ks) {
            const int koff = ks * 32;  // halfs

            // Two A fragments: A[m][k] = kv1[koff+k][(2pi+p)*16+m]
            Frag af0, af1;
            af0.h[0] = *(const v8h*)(abase + koff);
            af0.h[1] = *(const v8h*)(abase + koff + 16);
            af1.h[0] = *(const v8h*)(abase + koff + 16 * PADROW);
            af1.h[1] = *(const v8h*)(abase + koff + 16 * PADROW + 16);

#pragma unroll
            for (int t = 0; t < 4; ++t) {
                // B fragment: B[k][n] = kv2[koff+k][(4pj+t)*16+n] — reused twice
                Frag bf;
                bf.h[0] = *(const v8h*)(bbase + koff + t * 16 * PADROW);
                bf.h[1] = *(const v8h*)(bbase + koff + t * 16 * PADROW + 16);
                acc[0][t] = __builtin_amdgcn_wmma_f32_16x16x32_f16(
                    false, af0.v, false, bf.v, (short)0, acc[0][t], false, false);
                acc[1][t] = __builtin_amdgcn_wmma_f32_16x16x32_f16(
                    false, af1.v, false, bf.v, (short)0, acc[1][t], false, false);
            }
        }
        __syncthreads();
    }

    // Marginal partial sums (4 contributions per bin per block)
    atomicAdd(&marg1[bb * NBINS + bin], s1);
    atomicAdd(&marg2[bb * NBINS + bin], s2);

    // Joint partials: C/D layout -> lanes 0-15: M=r, N=lane; 16-31: M=r+8, N=lane-16
    float* jb = joint + bb * NBINS * NBINS;
    const int rbase = (lane >> 4) * 8;
    const int jcol  = lane & 15;
#pragma unroll
    for (int p = 0; p < 2; ++p) {
        const int ibase = (pi * 2 + p) * 16 + rbase;
#pragma unroll
        for (int t = 0; t < 4; ++t) {
            const int j = (pj * 4 + t) * 16 + jcol;
#pragma unroll
            for (int r = 0; r < 8; ++r) {
                atomicAdd(&jb[(ibase + r) * NBINS + j], acc[p][t][r]);
            }
        }
    }
}

// Grid: 8 blocks x 256 threads. Entropies + MI formula.
__global__ __launch_bounds__(256)
void mi_finalize(const float* __restrict__ joint, const float* __restrict__ marg1,
                 const float* __restrict__ marg2, float* __restrict__ out) {
    __shared__ float red[256];
    const int b = blockIdx.x, tid = threadIdx.x;

    auto reduce = [&](float v) -> float {
        red[tid] = v; __syncthreads();
        for (int s = 128; s > 0; s >>= 1) {
            if (tid < s) red[tid] += red[tid + s];
            __syncthreads();
        }
        float r = red[0]; __syncthreads();
        return r;
    };

    float m1   = marg1[b * NBINS + tid] * (1.0f / (float)NPIX);
    float sum1 = reduce(m1);
    float p1   = m1 / (sum1 + EPS);
    float H1   = reduce(-p1 * __log2f(p1 + EPS));

    float m2   = marg2[b * NBINS + tid] * (1.0f / (float)NPIX);
    float sum2 = reduce(m2);
    float p2   = m2 / (sum2 + EPS);
    float H2   = reduce(-p2 * __log2f(p2 + EPS));

    const float* jrow = joint + b * NBINS * NBINS + tid * NBINS;
    float s = 0.0f;
    for (int j = 0; j < NBINS; ++j) s += jrow[j];
    float jnorm = reduce(s) + EPS;

    float e = 0.0f;
    for (int j = 0; j < NBINS; ++j) {
        float p = jrow[j] / jnorm;
        e -= p * __log2f(p + EPS);
    }
    float H12 = reduce(e);

    if (tid == 0) out[b] = 2.0f * (H1 + H2 - H12) / (H1 + H2);
}

extern "C" void kernel_launch(void* const* d_in, const int* in_sizes, int n_in,
                              void* d_out, int out_size, void* d_ws, size_t ws_size,
                              hipStream_t stream) {
    (void)in_sizes; (void)n_in; (void)out_size; (void)ws_size;
    const float* in1  = (const float*)d_in[0];   // (4,1,256,256) f32
    const float* in2  = (const float*)d_in[1];   // (4,1,256,256) f32
    const float* bins = (const float*)d_in[2];   // (256,) f32
    float* ws    = (float*)d_ws;
    float* joint = ws;                           // 8 * 256 * 256
    float* marg1 = ws + 8 * NBINS * NBINS;       // 8 * 256
    float* marg2 = marg1 + 8 * NBINS;            // 8 * 256
    float* out   = (float*)d_out;                // (8,) f32

    const int nz = 8 * NBINS * NBINS + 2 * 8 * NBINS;
    mi_zero<<<(nz + 1023) / 1024, 1024, 0, stream>>>(ws, nz);
    mi_joint<<<dim3(NBLK, 8), 1024, 0, stream>>>(in1, in2, bins, joint, marg1, marg2);
    mi_finalize<<<8, 256, 0, stream>>>(joint, marg1, marg2, out);
}
